// GPT_71940702208679
// MI455X (gfx1250) — matrix operand order repaired and checked by reading, not compiled, not run on hardware
//
#include <hip/hip_runtime.h>

// ---------------------------------------------------------------------------
// CDNA5 (gfx1250) types + WMMA helper
// ---------------------------------------------------------------------------
typedef __attribute__((ext_vector_type(16))) __bf16 v16bf;
typedef __attribute__((ext_vector_type(4)))  __bf16 v4bf;
typedef __attribute__((ext_vector_type(8)))  float  v8f;
typedef __attribute__((ext_vector_type(4)))  int    v4i;

// Async global->LDS path (gfx1250 GLOBAL_LOAD_ASYNC_TO_LDS_*, ASYNCcnt),
// guarded so the file compiles on toolchains without the builtins.
#if defined(__has_builtin)
#if __has_builtin(__builtin_amdgcn_global_load_async_to_lds_b128) && \
    __has_builtin(__builtin_amdgcn_s_wait_asynccnt)
#define USE_ASYNC_LDS 1
#endif
#endif

#if defined(USE_ASYNC_LDS)
__device__ __forceinline__ void async_b128(const void* g, void* l) {
  // Builtin signature (from clang diagnostic): param0 is AS1 pointer to v4i.
  __builtin_amdgcn_global_load_async_to_lds_b128(
      (__attribute__((address_space(1))) v4i*)g,
      (__attribute__((address_space(3))) v4i*)l, /*offset=*/0, /*cpol=*/0);
}
#endif

__device__ __forceinline__ v8f wmma_bf16(v16bf a, v16bf b, v8f c) {
  // D = A(16x32 bf16) * B(32x16 bf16) + C(16x16 f32)
  return __builtin_amdgcn_wmma_f32_16x16x32_bf16(
      /*neg_a=*/false, a, /*neg_b=*/false, b,
      /*c_mod=*/(short)0, c, /*reuse_a=*/false, /*reuse_b=*/false);
}

// ---------------------------------------------------------------------------
// Fragment builders (CDNA5 wave32 WMMA register layouts, CDNA5 ISA)
// A 16x32 bf16: lane m=lane&15; VGPR v holds K pair k0 = 2v + 16*(lane>=16) (this code's contiguous k order, not the ISA table order)
// B 32x16 bf16: lane n=lane&15; VGPR v holds K pair k = 2v + 16*(lane>=16) (same contiguous order as A, not the ISA table)
// C/D 16x16 f32: lane n=lane&15; VGPR v holds row m = v + 8*(lane>=16)
// ---------------------------------------------------------------------------
__device__ __forceinline__ v16bf frag_a_row(const __bf16* src, int ld, int lane) {
  // A[m][k] = src[m*ld + k]
  int m = lane & 15, hk = (lane >> 4) * 16;
  v16bf a;
#pragma unroll
  for (int v = 0; v < 8; ++v) {
    int k0 = 2 * v + hk;
    a[2 * v]     = src[m * ld + k0];
    a[2 * v + 1] = src[m * ld + k0 + 1];
  }
  return a;
}

__device__ __forceinline__ v16bf frag_a_row_k16(const __bf16* src, int ld, int lane) {
  // A[m][k] = src[m*ld + k] for k<16, zero-padded to K=32
  int m = lane & 15, hk = (lane >> 4) * 16;
  v16bf a;
#pragma unroll
  for (int v = 0; v < 8; ++v) {
    int k0 = 2 * v + hk;
    bool ok = k0 < 16;
    a[2 * v]     = ok ? src[m * ld + k0]     : (__bf16)0.0f;
    a[2 * v + 1] = ok ? src[m * ld + k0 + 1] : (__bf16)0.0f;
  }
  return a;
}

__device__ __forceinline__ v16bf frag_a_col_k16(const __bf16* src, int ld, int col0, int lane) {
  // A[m][k] = src[k*ld + col0 + m] for k<16 (implicit transpose), zero-padded
  int m = lane & 15, hk = (lane >> 4) * 16;
  v16bf a;
#pragma unroll
  for (int v = 0; v < 8; ++v) {
    int k0 = 2 * v + hk;
    bool ok = k0 < 16;
    a[2 * v]     = ok ? src[k0 * ld + col0 + m]       : (__bf16)0.0f;
    a[2 * v + 1] = ok ? src[(k0 + 1) * ld + col0 + m] : (__bf16)0.0f;
  }
  return a;
}

__device__ __forceinline__ v16bf frag_b_rowT(const __bf16* src, int ld, int lane) {
  // B[k][n] = src[n*ld + k]  (source is row-major N x K, i.e. B = src^T)
  int n = lane & 15, kh = (lane >> 4) * 16;
  v16bf b;
#pragma unroll
  for (int v = 0; v < 8; ++v) {
    b[2 * v]     = src[n * ld + kh + 2 * v];
    b[2 * v + 1] = src[n * ld + kh + 2 * v + 1];
  }
  return b;
}

__device__ __forceinline__ v16bf frag_b_row_f32(const float* src, int ld, int lane) {
  // B[k][n] = (bf16)src[k*ld + n]  (f32 source, on-the-fly downconvert)
  int n = lane & 15, kh = (lane >> 4) * 16;
  v16bf b;
#pragma unroll
  for (int v = 0; v < 8; ++v) {
    b[2 * v]     = (__bf16)src[(kh + 2 * v) * ld + n];
    b[2 * v + 1] = (__bf16)src[(kh + 2 * v + 1) * ld + n];
  }
  return b;
}

__device__ __forceinline__ v16bf frag_b_row_k16(const __bf16* src, int ld, int col0, int lane) {
  // B[k][n] = src[k*ld + col0 + n] for k<16, zero-padded to K=32
  int n = lane & 15, kh = (lane >> 4) * 16;
  v16bf b;
#pragma unroll
  for (int v = 0; v < 8; ++v) {
    int k = kh + 2 * v;
    bool ok = k < 16;
    b[2 * v]     = ok ? src[k * ld + col0 + n]       : (__bf16)0.0f;
    b[2 * v + 1] = ok ? src[(k + 1) * ld + col0 + n] : (__bf16)0.0f;
  }
  return b;
}

// ---------------------------------------------------------------------------
// f32 -> bf16 conversion (memory-bound, trivially saturates HBM)
// ---------------------------------------------------------------------------
__global__ __launch_bounds__(256) void f32_to_bf16(const float* __restrict__ src,
                                                   __bf16* __restrict__ dst, int n4) {
  int i = blockIdx.x * blockDim.x + threadIdx.x;
  if (i < n4) {
    float4 f = reinterpret_cast<const float4*>(src)[i];
    v4bf o;
    o[0] = (__bf16)f.x; o[1] = (__bf16)f.y; o[2] = (__bf16)f.z; o[3] = (__bf16)f.w;
    reinterpret_cast<v4bf*>(dst)[i] = o;
  }
}

// ---------------------------------------------------------------------------
// GEMM: C[M][N] (f32) = A[M][K] (bf16 row-major) * W[N][K]^T (bf16 row-major)
// Block tile 64x128, K-step 32, 256 threads (8 waves), 4 WMMA tiles per wave.
// Async path: double-buffered LDS fill via GLOBAL_LOAD_ASYNC_TO_LDS_B128;
// s_wait_asynccnt 3 drains only the current tile's 3 in-flight ops (async
// loads complete in order), overlapping next-tile fill with WMMA.
// ---------------------------------------------------------------------------
#define BM 64
#define BN 128
#define BK 32

__global__ __launch_bounds__(256) void gemm_bf16_nt(const __bf16* __restrict__ A,
                                                    const __bf16* __restrict__ W,
                                                    float* __restrict__ C,
                                                    int M, int N, int K) {
#if defined(USE_ASYNC_LDS)
  __shared__ __align__(16) __bf16 As[2][BM][BK];
  __shared__ __align__(16) __bf16 Ws[2][BN][BK];
#else
  __shared__ __align__(16) __bf16 As[1][BM][BK];
  __shared__ __align__(16) __bf16 Ws[1][BN][BK];
#endif

  int tid  = threadIdx.x;
  int lane = tid & 31, wave = tid >> 5;
  int bm = blockIdx.y * BM, bn = blockIdx.x * BN;
  int wm = (wave >> 2) * 32;   // 0 / 32
  int wn = (wave & 3) * 32;    // 0 / 32 / 64 / 96

  v8f zf = {0.f, 0.f, 0.f, 0.f, 0.f, 0.f, 0.f, 0.f};
  v8f acc[2][2] = {{zf, zf}, {zf, zf}};

  int ra = tid >> 2, ca = (tid & 3) * 8;   // A-tile load coords (64 x 32)
  int rw = tid >> 1, cw = (tid & 1) * 16;  // W-tile load coords (128 x 32)

#if defined(USE_ASYNC_LDS)
  {
    const __bf16* ga = A + (size_t)(bm + ra) * K + ca;
    const __bf16* gw = W + (size_t)(bn + rw) * K + cw;
    async_b128(ga, &As[0][ra][ca]);
    async_b128(gw, &Ws[0][rw][cw]);
    async_b128(gw + 8, &Ws[0][rw][cw + 8]);
  }
  for (int k0 = 0; k0 < K; k0 += BK) {
    int buf = (k0 / BK) & 1;
    if (k0 + BK < K) {  // kick off next tile into the other buffer
      const __bf16* ga = A + (size_t)(bm + ra) * K + (k0 + BK) + ca;
      const __bf16* gw = W + (size_t)(bn + rw) * K + (k0 + BK) + cw;
      async_b128(ga, &As[buf ^ 1][ra][ca]);
      async_b128(gw, &Ws[buf ^ 1][rw][cw]);
      async_b128(gw + 8, &Ws[buf ^ 1][rw][cw + 8]);
      if (k0 + 2 * BK < K) __builtin_prefetch(ga + BK, 0, 1);
      __builtin_amdgcn_s_wait_asynccnt(3);  // current tile landed; next in flight
    } else {
      __builtin_amdgcn_s_wait_asynccnt(0);
    }
    __syncthreads();

    v16bf a0 = frag_a_row(&As[buf][wm][0],      BK, lane);
    v16bf a1 = frag_a_row(&As[buf][wm + 16][0], BK, lane);
    v16bf b0 = frag_b_rowT(&Ws[buf][wn][0],      BK, lane);
    v16bf b1 = frag_b_rowT(&Ws[buf][wn + 16][0], BK, lane);

    acc[0][0] = wmma_bf16(a0, b0, acc[0][0]);
    acc[0][1] = wmma_bf16(a0, b1, acc[0][1]);
    acc[1][0] = wmma_bf16(a1, b0, acc[1][0]);
    acc[1][1] = wmma_bf16(a1, b1, acc[1][1]);
    __syncthreads();  // tile fully consumed before it is refilled
  }
#else
  for (int k0 = 0; k0 < K; k0 += BK) {
    const __bf16* ga = A + (size_t)(bm + ra) * K + k0 + ca;
    *reinterpret_cast<uint4*>(&As[0][ra][ca]) = *reinterpret_cast<const uint4*>(ga);

    const __bf16* gw = W + (size_t)(bn + rw) * K + k0 + cw;
    *reinterpret_cast<uint4*>(&Ws[0][rw][cw])     = *reinterpret_cast<const uint4*>(gw);
    *reinterpret_cast<uint4*>(&Ws[0][rw][cw + 8]) = *reinterpret_cast<const uint4*>(gw + 8);

    if (k0 + BK < K) {  // pull the next K-slab toward L2 (global_prefetch_b8)
      __builtin_prefetch(ga + BK, 0, 1);
      __builtin_prefetch(gw + BK, 0, 1);
    }
    __syncthreads();

    v16bf a0 = frag_a_row(&As[0][wm][0],      BK, lane);
    v16bf a1 = frag_a_row(&As[0][wm + 16][0], BK, lane);
    v16bf b0 = frag_b_rowT(&Ws[0][wn][0],      BK, lane);
    v16bf b1 = frag_b_rowT(&Ws[0][wn + 16][0], BK, lane);

    acc[0][0] = wmma_bf16(a0, b0, acc[0][0]);
    acc[0][1] = wmma_bf16(a0, b1, acc[0][1]);
    acc[1][0] = wmma_bf16(a1, b0, acc[1][0]);
    acc[1][1] = wmma_bf16(a1, b1, acc[1][1]);
    __syncthreads();
  }
#endif

  int n = lane & 15, mh = (lane >> 4) * 8;
#pragma unroll
  for (int i = 0; i < 2; ++i)
#pragma unroll
    for (int j = 0; j < 2; ++j)
#pragma unroll
      for (int v = 0; v < 8; ++v) {
        int row = bm + wm + i * 16 + mh + v;
        int col = bn + wn + j * 16 + n;
        C[(size_t)row * N + col] = acc[i][j][v];
      }
}

// ---------------------------------------------------------------------------
// Chunked fastmax (linear) attention + RoPE.  One block per head (16 blocks),
// 256 threads = 8 waves.  Chunk = 16 tokens.  Per chunk:
//   inter:  out += vsum + Q @ S_prev            (waves 1-4, 2 WMMA each)
//   intra:  Sc = mask .* (1 + Q @ K^T)          (wave 0, 2 WMMA + rowsums)
//           out += Sc @ V                        (waves 0-3, 1 WMMA each)
//   state:  S += K^T @ V (rank-16)              (waves 4-7, 4 WMMA each)
//   den_i = t0 + q_i . ksum_prev + rowsum(Sc)_i  (folds arange + "+1" terms)
// D x D f32 state, ksum/vsum live in LDS.  K=16 fragments zero-padded to 32.
// ---------------------------------------------------------------------------
#define SLD 65  // padded stride of the 64x64 f32 state (avoids LDS bank conflicts)

__global__ __launch_bounds__(256) void fastmax_attn(const float* __restrict__ qkv,
                                                    const float* __restrict__ cosb,
                                                    const float* __restrict__ sinb,
                                                    __bf16* __restrict__ y) {
  __shared__ float S[64 * SLD];        // KV state (f32 accumulation)
  __shared__ float ksum[64], vsum[64];
  __shared__ float accN[16 * 64];      // inter-chunk numerator
  __shared__ float rs[16], deni[16];
  __shared__ __align__(4) __bf16 qc[16 * 64], kc[16 * 64], vc[16 * 64];
  __shared__ __align__(4) __bf16 Sc[16 * 16];  // masked (1 + QK^T) scores

  int tid = threadIdx.x, lane = tid & 31, wave = tid >> 5;
  int h = blockIdx.x;
  v8f zf = {0.f, 0.f, 0.f, 0.f, 0.f, 0.f, 0.f, 0.f};

  for (int i = tid; i < 64 * SLD; i += 256) S[i] = 0.f;
  if (tid < 64) { ksum[tid] = 0.f; vsum[tid] = 0.f; }
  __syncthreads();

  for (int chunk = 0; chunk < 128; ++chunk) {
    int t0 = chunk * 16;

    // ---- Phase L: load chunk, apply RoPE, downconvert to bf16 in LDS ----
    {
      int i  = tid >> 4;            // token within chunk
      int d0 = (tid & 15) * 4;      // 4 consecutive head dims
      int t  = t0 + i;
      const float* base = qkv + (size_t)t * 3072 + h * 192;  // [h][{q,k,v}][d]
#pragma unroll
      for (int u = 0; u < 4; ++u) {
        int d = d0 + u;
        float cs = cosb[t * 64 + d], sn = sinb[t * 64 + d];
        float qv = base[d], kv = base[64 + d], vv = base[128 + d];
        float qp, kp;
        if (d < 32) { qp = -base[d + 32];      kp = -base[64 + d + 32]; }
        else        { qp =  base[d - 32];      kp =  base[64 + d - 32]; }
        qc[i * 64 + d] = (__bf16)(qv * cs + qp * sn);
        kc[i * 64 + d] = (__bf16)(kv * cs + kp * sn);
        vc[i * 64 + d] = (__bf16)vv;
      }
    }
    __syncthreads();

    // ---- Phase 1 ----
    if (wave == 0) {
      // P = Q @ K^T (16x16, K=64 -> two WMMAs), then mask + "+1" + rowsums
      v8f p = zf;
      p = wmma_bf16(frag_a_row(qc, 64, lane),      frag_b_rowT(kc, 64, lane),      p);
      p = wmma_bf16(frag_a_row(qc + 32, 64, lane), frag_b_rowT(kc + 32, 64, lane), p);
      int j = lane & 15, mh = (lane >> 4) * 8;
#pragma unroll
      for (int v = 0; v < 8; ++v) {
        int m = mh + v;
        float val = (j <= m) ? (1.0f + p[v]) : 0.0f;  // causal mask + score offset
        Sc[m * 16 + j] = (__bf16)val;
        float s = val;
#pragma unroll
        for (int off = 1; off < 16; off <<= 1) s += __shfl_xor(s, off, 32);
        if (j == 0) rs[m] = s;   // (i_local+1) + sum_{j<=i} P[i][j]
      }
    } else if (wave <= 4) {
      // inter numerator: accN = vsum + Q @ S_prev   (one 16-wide N tile per wave)
      int nb = (wave - 1) * 16;
      int n = lane & 15, mh = (lane >> 4) * 8;
      v8f c;
#pragma unroll
      for (int v = 0; v < 8; ++v) c[v] = vsum[nb + n];
      c = wmma_bf16(frag_a_row(qc, 64, lane),      frag_b_row_f32(S + nb,            SLD, lane), c);
      c = wmma_bf16(frag_a_row(qc + 32, 64, lane), frag_b_row_f32(S + 32 * SLD + nb, SLD, lane), c);
#pragma unroll
      for (int v = 0; v < 8; ++v) accN[(mh + v) * 64 + nb + n] = c[v];
    } else if (wave == 5) {
      // inter denominator: deni_i = t0 + q_i . ksum_prev
      if (lane < 16) {
        float s = (float)t0;
        for (int d = 0; d < 64; ++d) s += (float)qc[lane * 64 + d] * ksum[d];
        deni[lane] = s;
      }
    }
    __syncthreads();

    // ---- Phase 2 ----
    if (wave < 4) {
      // out = (accN + Sc @ V) / den ; write y as bf16 for the proj GEMM
      int nb = wave * 16;
      int n = lane & 15, mh = (lane >> 4) * 8;
      v8f c;
#pragma unroll
      for (int v = 0; v < 8; ++v) c[v] = accN[(mh + v) * 64 + nb + n];
      c = wmma_bf16(frag_a_row_k16(Sc, 16, lane), frag_b_row_k16(vc, 64, nb, lane), c);
#pragma unroll
      for (int v = 0; v < 8; ++v) {
        int m = mh + v;
        float den = deni[m] + rs[m];
        y[(size_t)(t0 + m) * 1024 + h * 64 + nb + n] = (__bf16)(c[v] / den);
      }
    } else {
      // state update: S += K^T @ V  (rank-16, 16x16 tiles; 4 tiles per wave)
      int mb = (wave - 4) * 16;
      int n = lane & 15, mh = (lane >> 4) * 8;
      v16bf a = frag_a_col_k16(kc, 64, mb, lane);  // A[d][j] = K[j][mb+d]
#pragma unroll
      for (int nt = 0; nt < 4; ++nt) {
        int nb = nt * 16;
        v8f c;
#pragma unroll
        for (int v = 0; v < 8; ++v) c[v] = S[(mb + mh + v) * SLD + nb + n];
        c = wmma_bf16(a, frag_b_row_k16(vc, 64, nb, lane), c);
#pragma unroll
        for (int v = 0; v < 8; ++v) S[(mb + mh + v) * SLD + nb + n] = c[v];
      }
    }
    __syncthreads();

    // ---- Phase 3: fold chunk into ksum / vsum ----
    if (tid < 64) {
      float s = ksum[tid];
#pragma unroll
      for (int j = 0; j < 16; ++j) s += (float)kc[j * 64 + tid];
      ksum[tid] = s;
    } else if (tid < 128) {
      int e = tid - 64;
      float s = vsum[e];
#pragma unroll
      for (int j = 0; j < 16; ++j) s += (float)vc[j * 64 + e];
      vsum[e] = s;
    }
    __syncthreads();
  }
}

// ---------------------------------------------------------------------------
// Launch: convert -> QKV GEMM -> chunked fastmax attention -> proj GEMM
// ---------------------------------------------------------------------------
extern "C" void kernel_launch(void* const* d_in, const int* in_sizes, int n_in,
                              void* d_out, int out_size, void* d_ws, size_t ws_size,
                              hipStream_t stream) {
  (void)in_sizes; (void)n_in; (void)out_size; (void)ws_size;
  const float* x      = (const float*)d_in[0];  // 2048 x 1024
  const float* w_attn = (const float*)d_in[1];  // 3072 x 1024
  const float* w_proj = (const float*)d_in[2];  // 1024 x 1024
  const float* cosb   = (const float*)d_in[3];  // 2048 x 64
  const float* sinb   = (const float*)d_in[4];  // 2048 x 64
  float* out = (float*)d_out;                   // 2048 x 1024 f32

  // workspace layout (40 MiB total)
  __bf16* x_bf  = (__bf16*)d_ws;                       //  4 MiB
  __bf16* wa_bf = x_bf  + 2048 * 1024;                 //  6 MiB
  __bf16* wp_bf = wa_bf + 3072 * 1024;                 //  2 MiB
  float*  qkv   = (float*)(wp_bf + 1024 * 1024);       // 24 MiB
  __bf16* ybf   = (__bf16*)(qkv + 2048 * 3072);        //  4 MiB

  const int CT = 256;
  f32_to_bf16<<<(2048 * 1024 / 4 + CT - 1) / CT, CT, 0, stream>>>(x,      x_bf,  2048 * 1024 / 4);
  f32_to_bf16<<<(3072 * 1024 / 4 + CT - 1) / CT, CT, 0, stream>>>(w_attn, wa_bf, 3072 * 1024 / 4);
  f32_to_bf16<<<(1024 * 1024 / 4 + CT - 1) / CT, CT, 0, stream>>>(w_proj, wp_bf, 1024 * 1024 / 4);

  // qkv = x @ w_attn^T : M=2048, N=3072, K=1024
  gemm_bf16_nt<<<dim3(3072 / BN, 2048 / BM), 256, 0, stream>>>(x_bf, wa_bf, qkv, 2048, 3072, 1024);

  // RoPE + chunked fastmax attention, one block per head
  fastmax_attn<<<16, 256, 0, stream>>>(qkv, cosb, sinb, ybf);

  // out = y @ w_proj^T : M=2048, N=1024, K=1024
  gemm_bf16_nt<<<dim3(1024 / BN, 2048 / BM), 256, 0, stream>>>(ybf, wp_bf, out, 2048, 1024, 1024);
}
